// FprojLatentConditioned_21921513079373
// MI455X (gfx1250) — compile-verified, matrix operand order checked by reading, not run on hardware
//
#include <hip/hip_runtime.h>
#include <cmath>

// ---------------------------------------------------------------------------
// FprojLatentConditioned: per-sample 3x3 SVD (FP32 VALU) + 5-layer MLP on
// v_wmma_f32_16x16x32_f16.  Weights live in LDS as pre-packed B fragments.
// Each wave processes 32 samples (two 16-row M-tiles) per iteration; both
// M-tiles share each B fragment (2 WMMAs per ds wait).
// ---------------------------------------------------------------------------

typedef _Float16 h4   __attribute__((ext_vector_type(4)));
typedef _Float16 h8   __attribute__((ext_vector_type(8)));
typedef _Float16 v16h __attribute__((ext_vector_type(16)));
typedef float    v8f  __attribute__((ext_vector_type(8)));

#define ALPHA_C 0.1f

#define FRAG_BYTES   1280   // 32 lanes * 40B (32B payload + 8B pad -> conflict-free ds_load_b64)
#define LANE_STRIDE  40
#define NFRAGS       132    // W1:32  W2:32  W3:32  W4:32  W5:4
#define WBYTES       (NFRAGS * FRAG_BYTES)        // 168960
#define ROWB         272    // bytes per scratch row (128 halves + pad, 16B aligned)
#define WAVE_SCR     (32 * ROWB)                  // 8704 per wave per region
#define WAVES_PER_BLOCK 8
#define BLOCK_THREADS   (WAVES_PER_BLOCK * 32)
#define LDS_TOTAL    (WBYTES + 2 * WAVES_PER_BLOCK * WAVE_SCR)  // 308224 <= 320KB

#define W1_OFF 0
#define W2_OFF (32 * FRAG_BYTES)
#define W3_OFF (64 * FRAG_BYTES)
#define W4_OFF (96 * FRAG_BYTES)
#define W5_OFF (128 * FRAG_BYTES)

// ---------------------------------------------------------------------------
// Prep: pack fp32 weights into f16 WMMA B-fragments (K x N tiles of 32x16).
// ---------------------------------------------------------------------------
__global__ void pack_weights_f16(const float* __restrict__ W1, const float* __restrict__ W2,
                                 const float* __restrict__ W3, const float* __restrict__ W4,
                                 const float* __restrict__ W5, _Float16* __restrict__ wp)
{
    int gid = blockIdx.x * blockDim.x + threadIdx.x;
    if (gid >= NFRAGS * 32) return;
    int frag = gid >> 5;
    int lane = gid & 31;

    const float* W; int K, Nact, fi;
    if      (frag < 32)  { W = W1; K = 94;  Nact = 128; fi = frag;       }
    else if (frag < 64)  { W = W2; K = 128; Nact = 128; fi = frag - 32;  }
    else if (frag < 96)  { W = W3; K = 128; Nact = 128; fi = frag - 64;  }
    else if (frag < 128) { W = W4; K = 128; Nact = 128; fi = frag - 96;  }
    else                 { W = W5; K = 128; Nact = 9;   fi = frag - 128; }

    int nt = fi >> 2, kt = fi & 3;
    int n  = nt * 16 + (lane & 15);
    int kb = kt * 32 + (lane >> 4) * 8;
    _Float16* dst = wp + (frag * FRAG_BYTES + lane * LANE_STRIDE) / 2;
#pragma unroll
    for (int j = 0; j < 8; ++j) {
        int k0 = kb + j, k1 = kb + 16 + j;
        float v0 = (k0 < K && n < Nact) ? W[k0 * Nact + n] : 0.0f;
        float v1 = (k1 < K && n < Nact) ? W[k1 * Nact + n] : 0.0f;
        dst[j]     = (_Float16)v0;
        dst[8 + j] = (_Float16)v1;
    }
}

// ---------------------------------------------------------------------------
// Branchless exact-erf GELU (Abramowitz-Stegun 7.1.26, |eps| <= 1.5e-7).
// Uses v_rcp_f32 + v_exp_f32; no EXEC divergence.
// ---------------------------------------------------------------------------
__device__ __forceinline__ float gelu_fast(float x) {
    float z  = x * 0.70710678118654752f;
    float az = fabsf(z);
    float t  = __builtin_amdgcn_rcpf(1.0f + 0.3275911f * az);
    float p  = t * (0.254829592f +
               t * (-0.284496736f +
               t * (1.421413741f +
               t * (-1.453152027f +
               t * 1.061405429f))));
    float e  = __builtin_amdgcn_exp2f(-1.4426950408889634f * az * az);
    float erfv = copysignf(1.0f - p * e, z);
    return 0.5f * x * (1.0f + erfv);
}

// A fragment (16x32 f16): lane holds row (lane&15); hi selects K halves.
__device__ __forceinline__ v16h load_a_frag(const char* rowBase, int kt, int hi) {
    const h8* p0 = (const h8*)(rowBase + kt * 64 + hi * 16);       // halves kb..kb+7
    const h8* p1 = (const h8*)(rowBase + kt * 64 + hi * 16 + 32);  // halves kb+16..kb+23
    h8 x0 = *p0, x1 = *p1;
    v16h a;
#pragma unroll
    for (int j = 0; j < 8; ++j) { a[j] = x0[j]; a[8 + j] = x1[j]; }
    return a;
}

// B fragment: conflict-free ds_load_b64 x4 from the padded packed layout.
__device__ __forceinline__ v16h load_b_frag(const char* wbase, int fragIdx, int lane) {
    const char* p = wbase + fragIdx * FRAG_BYTES + lane * LANE_STRIDE;
    h4 b0 = *(const h4*)(p);
    h4 b1 = *(const h4*)(p + 8);
    h4 b2 = *(const h4*)(p + 16);
    h4 b3 = *(const h4*)(p + 24);
    v16h b;
#pragma unroll
    for (int j = 0; j < 4; ++j) {
        b[j] = b0[j]; b[4 + j] = b1[j]; b[8 + j] = b2[j]; b[12 + j] = b3[j];
    }
    return b;
}

#define WMMA_F16(A, B, C) \
    __builtin_amdgcn_wmma_f32_16x16x32_f16(false, (A), false, (B), (short)0, (C), false, false)

// One hidden layer over all 32 rows: D(32x128) = gelu(A(32x128) x W + b)
__device__ __forceinline__ void run_hidden_layer32(const char* aRegion, char* dRegion,
                                                   const char* wfrag, const float* bv, // bv[8] in regs
                                                   int lane)
{
    const int hi  = lane >> 4;
    const int nlo = lane & 15;
    asm volatile("s_wait_dscnt 0" ::: "memory");   // prior wave-local LDS stores visible
    const char* row0 = aRegion + nlo * ROWB;
    const char* row1 = aRegion + (16 + nlo) * ROWB;
    v16h a00 = load_a_frag(row0, 0, hi), a01 = load_a_frag(row0, 1, hi);
    v16h a02 = load_a_frag(row0, 2, hi), a03 = load_a_frag(row0, 3, hi);
    v16h a10 = load_a_frag(row1, 0, hi), a11 = load_a_frag(row1, 1, hi);
    v16h a12 = load_a_frag(row1, 2, hi), a13 = load_a_frag(row1, 3, hi);
    for (int nt = 0; nt < 8; ++nt) {
        float bvv = bv[nt];
        v8f c0, c1;
#pragma unroll
        for (int i = 0; i < 8; ++i) { c0[i] = bvv; c1[i] = bvv; }
        v16h b0 = load_b_frag(wfrag, nt * 4 + 0, lane);
        v16h b1 = load_b_frag(wfrag, nt * 4 + 1, lane);
        c0 = WMMA_F16(a00, b0, c0);  c1 = WMMA_F16(a10, b0, c1);
        v16h b2 = load_b_frag(wfrag, nt * 4 + 2, lane);
        c0 = WMMA_F16(a01, b1, c0);  c1 = WMMA_F16(a11, b1, c1);
        v16h b3 = load_b_frag(wfrag, nt * 4 + 3, lane);
        c0 = WMMA_F16(a02, b2, c0);  c1 = WMMA_F16(a12, b2, c1);
        c0 = WMMA_F16(a03, b3, c0);  c1 = WMMA_F16(a13, b3, c1);
#pragma unroll
        for (int v = 0; v < 8; ++v) {
            int r0 = v + 8 * hi;           // C layout: VGPR v -> M = v + 8*(lane>=16)
            float g0 = gelu_fast(c0[v]);
            float g1 = gelu_fast(c1[v]);
            ((_Float16*)(dRegion + r0 * ROWB))[nt * 16 + nlo]        = (_Float16)g0;
            ((_Float16*)(dRegion + (16 + r0) * ROWB))[nt * 16 + nlo] = (_Float16)g1;
        }
    }
}

// Final layer over all 32 rows: x(32x16, n<9 meaningful) = A x W5 + b5, stored f32.
__device__ __forceinline__ void run_final_layer32(const char* aRegion, char* xRegion,
                                                  const char* wfrag, float b5v, int lane)
{
    const int hi  = lane >> 4;
    const int nlo = lane & 15;
    asm volatile("s_wait_dscnt 0" ::: "memory");
    const char* row0 = aRegion + nlo * ROWB;
    const char* row1 = aRegion + (16 + nlo) * ROWB;
    v8f c0, c1;
#pragma unroll
    for (int i = 0; i < 8; ++i) { c0[i] = b5v; c1[i] = b5v; }
#pragma unroll
    for (int kt = 0; kt < 4; ++kt) {
        v16h b = load_b_frag(wfrag, kt, lane);
        c0 = WMMA_F16(load_a_frag(row0, kt, hi), b, c0);
        c1 = WMMA_F16(load_a_frag(row1, kt, hi), b, c1);
    }
#pragma unroll
    for (int v = 0; v < 8; ++v) {
        int r0 = v + 8 * hi;
        ((float*)(xRegion + r0 * ROWB))[nlo]        = c0[v];
        ((float*)(xRegion + (16 + r0) * ROWB))[nlo] = c1[v];
    }
}

// ---------------------------------------------------------------------------
// Main kernel
// ---------------------------------------------------------------------------
__global__ void __launch_bounds__(BLOCK_THREADS, 1)
fproj_mlp_kernel(const float* __restrict__ Fm, const float* __restrict__ latw,
                 const float* __restrict__ b1, const float* __restrict__ b2,
                 const float* __restrict__ b3, const float* __restrict__ b4,
                 const float* __restrict__ b5, const int* __restrict__ traj,
                 const _Float16* __restrict__ wp, float* __restrict__ out, int nB)
{
    extern __shared__ char smem[];
    const int tid  = threadIdx.x;
    const int wave = tid >> 5;
    const int lane = tid & 31;
    const int nlo  = lane & 15;

    // ---- cooperative copy: packed weights -> LDS (once per block) ----
    {
        const float4* src = (const float4*)wp;
        float4* dst = (float4*)smem;
        for (int i = tid; i < WBYTES / 16; i += BLOCK_THREADS) dst[i] = src[i];
    }
    __syncthreads();

    char* featScr = smem + WBYTES + wave * WAVE_SCR;                              // 32 x f16[128] rows
    char* hScr    = smem + WBYTES + WAVES_PER_BLOCK * WAVE_SCR + wave * WAVE_SCR; // 32 x f16[128] rows

    // ---- hoist biases into registers (loop invariant, per-lane column nlo) ----
    float bL1[8], bL2[8], bL3[8], bL4[8];
#pragma unroll
    for (int nt = 0; nt < 8; ++nt) {
        bL1[nt] = b1[nt * 16 + nlo];
        bL2[nt] = b2[nt * 16 + nlo];
        bL3[nt] = b3[nt * 16 + nlo];
        bL4[nt] = b4[nt * 16 + nlo];
    }
    const float b5v = (nlo < 9) ? b5[nlo] : 0.0f;

    // ---- one-time constant feature columns: latent[64] + zero pad ----
    {
        _Float16* row = (_Float16*)(featScr + lane * ROWB);
        const float* lp = latw + (long)traj[0] * 64;
#pragma unroll
        for (int j = 0; j < 64; ++j) row[30 + j] = (_Float16)lp[j];
#pragma unroll
        for (int k = 94; k < 128; ++k) row[k] = (_Float16)0.0f;
    }

    const int nTiles = (nB + 31) >> 5;
    const int gw  = blockIdx.x * WAVES_PER_BLOCK + wave;
    const int gws = gridDim.x * WAVES_PER_BLOCK;

    for (int t = gw; t < nTiles; t += gws) {
        const int  s     = t * 32 + lane;
        const bool valid = (s < nB);

        // prefetch next tile's F rows into cache
        {
            long ns = (long)(t + gws) * 32 + lane;
            if (ns < (long)nB) __builtin_prefetch(Fm + ns * 9, 0, 1);
        }

        float f[9];
#pragma unroll
        for (int i = 0; i < 9; ++i) f[i] = 0.0f;
        f[0] = f[4] = f[8] = 1.0f;
        if (valid) {
#pragma unroll
            for (int i = 0; i < 9; ++i) f[i] = Fm[(long)s * 9 + i];
        }

        // ---- 3x3 SVD via cyclic Jacobi on S = F^T F ----
        float S[3][3], V[3][3];
#pragma unroll
        for (int i = 0; i < 3; ++i)
#pragma unroll
            for (int j = 0; j < 3; ++j) {
                float acc = 0.0f;
#pragma unroll
                for (int k = 0; k < 3; ++k) acc += f[k * 3 + i] * f[k * 3 + j];
                S[i][j] = acc;
                V[i][j] = (i == j) ? 1.0f : 0.0f;
            }
#pragma unroll
        for (int sweep = 0; sweep < 4; ++sweep) {
#pragma unroll
            for (int pq = 0; pq < 3; ++pq) {
                const int p = (pq == 2) ? 1 : 0;
                const int q = (pq == 0) ? 1 : 2;
                float apq = S[p][q];
                float c = 1.0f, sn = 0.0f;
                if (fabsf(apq) > 1e-12f) {
                    float tau = (S[q][q] - S[p][p]) / (2.0f * apq);
                    float tt  = ((tau >= 0.0f) ? 1.0f : -1.0f) /
                                (fabsf(tau) + sqrtf(1.0f + tau * tau));
                    c  = 1.0f / sqrtf(1.0f + tt * tt);
                    sn = tt * c;
                }
#pragma unroll
                for (int k = 0; k < 3; ++k) {
                    float skp = S[k][p], skq = S[k][q];
                    S[k][p] = c * skp - sn * skq;
                    S[k][q] = sn * skp + c * skq;
                }
#pragma unroll
                for (int k = 0; k < 3; ++k) {
                    float spk = S[p][k], sqk = S[q][k];
                    S[p][k] = c * spk - sn * sqk;
                    S[q][k] = sn * spk + c * sqk;
                }
#pragma unroll
                for (int k = 0; k < 3; ++k) {
                    float vkp = V[k][p], vkq = V[k][q];
                    V[k][p] = c * vkp - sn * vkq;
                    V[k][q] = sn * vkp + c * vkq;
                }
            }
        }
        float d0 = S[0][0], d1 = S[1][1], d2 = S[2][2];
        if (d0 < d1) { float tv = d0; d0 = d1; d1 = tv;
#pragma unroll
            for (int k = 0; k < 3; ++k) { float tc = V[k][0]; V[k][0] = V[k][1]; V[k][1] = tc; } }
        if (d0 < d2) { float tv = d0; d0 = d2; d2 = tv;
#pragma unroll
            for (int k = 0; k < 3; ++k) { float tc = V[k][0]; V[k][0] = V[k][2]; V[k][2] = tc; } }
        if (d1 < d2) { float tv = d1; d1 = d2; d2 = tv;
#pragma unroll
            for (int k = 0; k < 3; ++k) { float tc = V[k][1]; V[k][1] = V[k][2]; V[k][2] = tc; } }
        float sg[3] = { sqrtf(fmaxf(d0, 0.0f)), sqrtf(fmaxf(d1, 0.0f)), sqrtf(fmaxf(d2, 0.0f)) };

        float U[3][3];
#pragma unroll
        for (int j = 0; j < 3; ++j) {
            float inv = 1.0f / fmaxf(sg[j], 1e-20f);
#pragma unroll
            for (int i = 0; i < 3; ++i) {
                float acc = 0.0f;
#pragma unroll
                for (int k = 0; k < 3; ++k) acc += f[i * 3 + k] * V[k][j];
                U[i][j] = acc * inv;
            }
        }
        float R[9];
#pragma unroll
        for (int i = 0; i < 3; ++i)
#pragma unroll
            for (int j = 0; j < 3; ++j) {
                float acc = 0.0f;
#pragma unroll
                for (int k = 0; k < 3; ++k) acc += U[i][k] * V[j][k];   // R = U @ V^T
                R[i * 3 + j] = acc;
            }

        // ---- feature row (f16): [F(9), U(9), sigma(3), Vh(9)] ----
        {
            _Float16* row = (_Float16*)(featScr + lane * ROWB);
#pragma unroll
            for (int i = 0; i < 9; ++i) row[i] = (_Float16)f[i];
#pragma unroll
            for (int i = 0; i < 3; ++i)
#pragma unroll
                for (int j = 0; j < 3; ++j) row[9 + i * 3 + j] = (_Float16)U[i][j];
            row[18] = (_Float16)sg[0]; row[19] = (_Float16)sg[1]; row[20] = (_Float16)sg[2];
#pragma unroll
            for (int i = 0; i < 3; ++i)
#pragma unroll
                for (int j = 0; j < 3; ++j) row[21 + i * 3 + j] = (_Float16)V[j][i]; // Vh
        }

        // ---- WMMA MLP over all 32 rows at once ----
        run_hidden_layer32(featScr, hScr, (const char*)smem + W1_OFF, bL1, lane);
        run_hidden_layer32(hScr,    hScr, (const char*)smem + W2_OFF, bL2, lane);
        run_hidden_layer32(hScr,    hScr, (const char*)smem + W3_OFF, bL3, lane);
        run_hidden_layer32(hScr,    hScr, (const char*)smem + W4_OFF, bL4, lane);
        run_final_layer32 (hScr,    hScr, (const char*)smem + W5_OFF, b5v, lane);
        asm volatile("s_wait_dscnt 0" ::: "memory");

        if (valid) {
            const float* xr = (const float*)(hScr + lane * ROWB);  // row == lane
            float xs[3][3];
#pragma unroll
            for (int i = 0; i < 3; ++i)
#pragma unroll
                for (int j = 0; j < 3; ++j)
                    xs[i][j] = 0.5f * (xr[i * 3 + j] + xr[j * 3 + i]);
#pragma unroll
            for (int i = 0; i < 3; ++i)
#pragma unroll
                for (int j = 0; j < 3; ++j) {
                    float acc = 0.0f;
#pragma unroll
                    for (int k = 0; k < 3; ++k) acc += R[i * 3 + k] * xs[k][j];
                    out[(long)s * 9 + i * 3 + j] = ALPHA_C * acc + f[i * 3 + j];
                }
        }
    }
}

// ---------------------------------------------------------------------------
// Launch
// ---------------------------------------------------------------------------
extern "C" void kernel_launch(void* const* d_in, const int* in_sizes, int n_in,
                              void* d_out, int out_size, void* d_ws, size_t ws_size,
                              hipStream_t stream)
{
    const float* F    = (const float*)d_in[0];
    const float* latw = (const float*)d_in[1];
    const float* W1 = (const float*)d_in[2];  const float* b1 = (const float*)d_in[3];
    const float* W2 = (const float*)d_in[4];  const float* b2 = (const float*)d_in[5];
    const float* W3 = (const float*)d_in[6];  const float* b3 = (const float*)d_in[7];
    const float* W4 = (const float*)d_in[8];  const float* b4 = (const float*)d_in[9];
    const float* W5 = (const float*)d_in[10]; const float* b5 = (const float*)d_in[11];
    const int* traj = (const int*)d_in[12];
    float* out = (float*)d_out;
    const int nB = in_sizes[0] / 9;

    _Float16* wp = (_Float16*)d_ws;
    pack_weights_f16<<<(NFRAGS * 32 + 255) / 256, 256, 0, stream>>>(W1, W2, W3, W4, W5, wp);

    const int nTiles = (nB + 31) / 32;
    int blocks = (nTiles + WAVES_PER_BLOCK - 1) / WAVES_PER_BLOCK;
    if (blocks > 1024) blocks = 1024;
    if (blocks < 1)    blocks = 1;

    static_assert(LDS_TOTAL <= 320 * 1024, "LDS budget exceeded");
    (void)hipFuncSetAttribute((const void*)fproj_mlp_kernel,
                              hipFuncAttributeMaxDynamicSharedMemorySize, LDS_TOTAL);
    fproj_mlp_kernel<<<blocks, BLOCK_THREADS, LDS_TOTAL, stream>>>(
        F, latw, b1, b2, b3, b4, b5, traj, wp, out, nB);
}
